// SelfMatcher_6725918785797
// MI455X (gfx1250) — matrix operand order repaired and checked by reading, not compiled, not run on hardware
//
#include <hip/hip_runtime.h>

// ---------------------------------------------------------------------------
// Problem constants (reference: L=1024, B=24, D=512)
// ---------------------------------------------------------------------------
static constexpr int L = 1024;
static constexpr int B = 24;
static constexpr int D = 512;
static constexpr int D3 = 3 * D;

typedef __attribute__((ext_vector_type(16))) __bf16 v16bf;
typedef __attribute__((ext_vector_type(8)))  float  v8f;

// ---------------------------------------------------------------------------
// Fast transcendental helpers (v_exp_f32 / v_rcp_f32)
// ---------------------------------------------------------------------------
__device__ __forceinline__ float fast_exp2(float x) { return __builtin_amdgcn_exp2f(x); }
__device__ __forceinline__ float fast_rcp(float x)  { return __builtin_amdgcn_rcpf(x); }

__device__ __forceinline__ float fast_tanh(float x) {
  // tanh(x) = 1 - 2/(exp(2x)+1), exp(2x) = exp2(x * 2*log2(e))
  float e = fast_exp2(x * 2.8853900817779268f);
  return 1.0f - 2.0f * fast_rcp(e + 1.0f);
}
__device__ __forceinline__ float fast_sigmoid(float x) {
  return fast_rcp(1.0f + fast_exp2(x * -1.4426950408889634f));
}

// ---------------------------------------------------------------------------
// Async global->LDS copy (CDNA5 ASYNCcnt path).  Copies 16 bytes per lane.
// ldsoff = low 32 bits of the generic shared-memory address.
// ---------------------------------------------------------------------------
__device__ __forceinline__ void async_copy_b128(void* lds_dst, const void* gsrc) {
  unsigned ldsoff = (unsigned)(size_t)lds_dst;
  asm volatile("global_load_async_to_lds_b128 %0, %1, off"
               :: "v"(ldsoff), "v"(gsrc) : "memory");
}
__device__ __forceinline__ void wait_asynccnt0() {
  asm volatile("s_wait_asynccnt 0x0" ::: "memory");
}

// ---------------------------------------------------------------------------
// WMMA helpers: bf16 16x16x32, f32 accumulate.
// Fragment layouts used here (wave32; A/B k order is contiguous, not the ISA table order):
//  A (16x32 MxK): lane m=l&15; elem e -> K = e + (l<16?0:16)
//  B (32x16 KxN): lane n=l&15; elem e -> K = e + (l<16?0:16)
//  C/D (16x16):   lane n=l&15; vgpr j -> M = j + (l<16?0:8)
// ---------------------------------------------------------------------------
__device__ __forceinline__ v8f wmma_bf16(v16bf a, v16bf b, v8f c) {
  return __builtin_amdgcn_wmma_f32_16x16x32_bf16(false, a, false, b, (short)0, c,
                                                 false, false);
}

// A fragment from row-major [m][k] storage (leading dim ld)
__device__ __forceinline__ v16bf frag_a_rm(const __bf16* p, int ld) {
  const int lane = threadIdx.x & 31;
  const int m = lane & 15;
  const int kb = (lane < 16) ? 0 : 16;
  v16bf f;
#pragma unroll
  for (int e = 0; e < 16; ++e) {
    const int k = kb + e;
    f[e] = p[m * ld + k];
  }
  return f;
}

// B fragment from N-major [n][k] storage (i.e. W row-major rows=n, cols=k)
__device__ __forceinline__ v16bf frag_b_nm(const __bf16* p, int ld) {
  const int lane = threadIdx.x & 31;
  const int n = lane & 15;
  const int kb = (lane < 16) ? 0 : 16;
  v16bf f;
#pragma unroll
  for (int e = 0; e < 16; ++e) f[e] = p[n * ld + kb + e];
  return f;
}

// B fragment from K-major [k][n] storage
__device__ __forceinline__ v16bf frag_b_km(const __bf16* p, int ld) {
  const int lane = threadIdx.x & 31;
  const int n = lane & 15;
  const int kb = (lane < 16) ? 0 : 16;
  v16bf f;
#pragma unroll
  for (int e = 0; e < 16; ++e) f[e] = p[(kb + e) * ld + n];
  return f;
}

// ---------------------------------------------------------------------------
// Kernel 1: generic Y[M,N] = X[M,K] * W[N,K]^T (+bias[N]), all f32 in memory,
// bf16 WMMA compute with f32 accumulation.  WG tile 128x64, 8 waves (2x2 WMMA
// tiles each), K-step 32.  Requires M%128==0, N%64==0, K%32==0.
// ---------------------------------------------------------------------------
__global__ __launch_bounds__(256) void gemm_nt_bf16(
    const float* __restrict__ X, const float* __restrict__ W,
    const float* __restrict__ bias, float* __restrict__ Y,
    int M, int N, int K) {
  __shared__ __bf16 As[128][32];  // 8 KB
  __shared__ __bf16 Bs[64][32];   // 4 KB
  const int tid = threadIdx.x;
  const int wave = tid >> 5;
  const int lane = tid & 31;
  const int m0 = blockIdx.x * 128;
  const int n0 = blockIdx.y * 64;
  const int wr = wave >> 1;  // 0..3 -> 32-row strip
  const int wc = wave & 1;   // 0..1 -> 32-col strip
  v8f acc[2][2] = {};

  for (int k0 = 0; k0 < K; k0 += 32) {
    // prefetch next K tile (one prefetch per row covers the 128B row-chunk)
    if (k0 + 32 < K) {
      if (tid < 128) __builtin_prefetch(&X[(size_t)(m0 + tid) * K + (k0 + 32)], 0, 1);
      else if (tid < 192)
        __builtin_prefetch(&W[(size_t)(n0 + (tid - 128) % 64) * K + (k0 + 32)], 0, 1);
    }
    for (int idx = tid; idx < 128 * 32; idx += 256) {
      const int r = idx >> 5, c = idx & 31;
      As[r][c] = (__bf16)X[(size_t)(m0 + r) * K + (k0 + c)];
    }
    for (int idx = tid; idx < 64 * 32; idx += 256) {
      const int r = idx >> 5, c = idx & 31;
      Bs[r][c] = (__bf16)W[(size_t)(n0 + r) * K + (k0 + c)];
    }
    __syncthreads();
#pragma unroll
    for (int mi = 0; mi < 2; ++mi) {
      const v16bf af = frag_a_rm(&As[wr * 32 + mi * 16][0], 32);
#pragma unroll
      for (int ni = 0; ni < 2; ++ni) {
        const v16bf bfr = frag_b_nm(&Bs[wc * 32 + ni * 16][0], 32);
        acc[mi][ni] = wmma_bf16(af, bfr, acc[mi][ni]);
      }
    }
    __syncthreads();
  }

  const int cn = lane & 15;
  const int rb = (lane < 16) ? 0 : 8;
#pragma unroll
  for (int mi = 0; mi < 2; ++mi) {
#pragma unroll
    for (int ni = 0; ni < 2; ++ni) {
      const int gn = n0 + wc * 32 + ni * 16 + cn;
      const float bv = bias ? bias[gn] : 0.0f;
#pragma unroll
      for (int j = 0; j < 8; ++j) {
        const int gm = m0 + wr * 32 + mi * 16 + rb + j;
        Y[(size_t)gm * N + gn] = acc[mi][ni][j] + bv;
      }
    }
  }
}

// ---------------------------------------------------------------------------
// Kernel 2: additive-attention scores + softmax -> bf16 probs.
// Grid: (B, L/8).  One WG handles 8 query positions for one batch b.
// s[i,l] = sum_h tanh(q[i,h] + k[l,h]) * V[b,h]   (pure VALU/trans work)
// P[b][i][l] = softmax_l(s[i,:])                  (bf16 to global)
// LDS: 16 + 2 + 32 = 50 KB.
// ---------------------------------------------------------------------------
__global__ __launch_bounds__(256) void scores_kernel(
    const float* __restrict__ Wpv,   // [L,B,D] query-side projection
    const float* __restrict__ Wpvp,  // [L,B,D] key-side projection
    const float* __restrict__ Vv,    // [B,D]
    __bf16* __restrict__ P) {        // [B][L][L]
  constexpr int TI = 8;
  const int b = blockIdx.x;
  const int i0 = blockIdx.y * TI;
  __shared__ float s_q[TI][D];   // 16 KB
  __shared__ float s_V[D];       //  2 KB
  __shared__ float s_sc[TI][L];  // 32 KB
  const int tid = threadIdx.x, wave = tid >> 5, lane = tid & 31;

  for (int idx = tid; idx < TI * D; idx += 256) {
    const int r = idx >> 9, c = idx & (D - 1);
    s_q[r][c] = Wpv[((size_t)(i0 + r) * B + b) * D + c];
  }
  for (int idx = tid; idx < D; idx += 256) s_V[idx] = Vv[b * D + idx];
  __syncthreads();

  // --- score rows: wave handles keys l = wave, wave+8, ...
  for (int l = wave; l < L; l += 8) {
    // prefetch next key row for this wave (each lane covers 64B)
    if (l + 8 < L)
      __builtin_prefetch(Wpvp + ((size_t)(l + 8) * B + b) * D + lane * 16, 0, 1);
    float acc[TI];
#pragma unroll
    for (int i = 0; i < TI; ++i) acc[i] = 0.0f;
    const float* kp = Wpvp + ((size_t)l * B + b) * D;
    for (int hc = 0; hc < D; hc += 32) {
      const int h = hc + lane;
      const float kv = kp[h];
      const float vv = s_V[h];
#pragma unroll
      for (int i = 0; i < TI; ++i)
        acc[i] += fast_tanh(s_q[i][h] + kv) * vv;
    }
#pragma unroll
    for (int i = 0; i < TI; ++i) {
      float a = acc[i];
      a += __shfl_xor(a, 16);
      a += __shfl_xor(a, 8);
      a += __shfl_xor(a, 4);
      a += __shfl_xor(a, 2);
      a += __shfl_xor(a, 1);
      if (lane == i) s_sc[i][l] = a;
    }
  }
  __syncthreads();

  // --- softmax over l, one row per wave (TI == #waves)
  {
    const int i = wave;
    float mx = -3.402823466e38f;
    for (int l = lane; l < L; l += 32) mx = fmaxf(mx, s_sc[i][l]);
    mx = fmaxf(mx, __shfl_xor(mx, 16));
    mx = fmaxf(mx, __shfl_xor(mx, 8));
    mx = fmaxf(mx, __shfl_xor(mx, 4));
    mx = fmaxf(mx, __shfl_xor(mx, 2));
    mx = fmaxf(mx, __shfl_xor(mx, 1));
    float sum = 0.0f;
    for (int l = lane; l < L; l += 32) {
      const float p = fast_exp2((s_sc[i][l] - mx) * 1.4426950408889634f);
      s_sc[i][l] = p;
      sum += p;
    }
    sum += __shfl_xor(sum, 16);
    sum += __shfl_xor(sum, 8);
    sum += __shfl_xor(sum, 4);
    sum += __shfl_xor(sum, 2);
    sum += __shfl_xor(sum, 1);
    const float inv = fast_rcp(sum);
    __bf16* Prow = P + ((size_t)b * L + (i0 + i)) * L;
    for (int l = lane; l < L; l += 32) Prow[l] = (__bf16)(s_sc[i][l] * inv);
  }
}

// ---------------------------------------------------------------------------
// Kernel 3: batched context GEMM.  Per b:  ctx[i,d] = sum_l P[b][i][l]*v[l,b,d]
// Grid: (B, L/64, D/64).  WG tile 64x64, K-step 32, WMMA.
// A tile (bf16 probs) is staged with the CDNA5 async global->LDS engine:
// 64x32 bf16 = 4 KB = exactly one b128 per thread.
// ---------------------------------------------------------------------------
__global__ __launch_bounds__(256) void ctx_gemm_kernel(
    const __bf16* __restrict__ P,  // [B][L][L]
    const float* __restrict__ v,   // [L,B,D]
    float* __restrict__ ctx) {     // [L,B,D]
  const int b = blockIdx.x;
  const int i0 = blockIdx.y * 64;
  const int d0 = blockIdx.z * 64;
  __shared__ __bf16 As[64][32];  // 4 KB, probs [i][l]
  __shared__ __bf16 Bs[32][64];  // 4 KB, v tile [l][d] (K-major)
  const int tid = threadIdx.x, wave = tid >> 5, lane = tid & 31;
  const int mr = wave & 3;   // 0..3 m-tile
  const int nc = wave >> 2;  // 0..1 -> two n-tiles
  v8f acc[2] = {};
  const __bf16* Pb = P + (size_t)b * L * L;
  // per-thread async-copy coordinates: row ar (0..63), 8-elem group aq (0..3)
  const int ar = tid >> 2;
  const int aq = (tid & 3) * 8;

  for (int k0 = 0; k0 < L; k0 += 32) {
    // A tile: async DMA copy, bf16 -> bf16 verbatim (16 B per thread)
    async_copy_b128(&As[ar][aq], Pb + (size_t)(i0 + ar) * L + k0 + aq);
    // prefetch next v K-tile (one per row)
    if (k0 + 32 < L && tid < 32)
      __builtin_prefetch(&v[((size_t)(k0 + 32 + tid) * B + b) * D + d0], 0, 1);
    // B tile: f32 -> bf16 conversion staging
    for (int idx = tid; idx < 32 * 64; idx += 256) {
      const int r = idx >> 6, c = idx & 63;
      Bs[r][c] = (__bf16)v[((size_t)(k0 + r) * B + b) * D + (d0 + c)];
    }
    wait_asynccnt0();
    __syncthreads();
    const v16bf af = frag_a_rm(&As[mr * 16][0], 32);
#pragma unroll
    for (int t = 0; t < 2; ++t) {
      const v16bf bfr = frag_b_km(&Bs[0][(nc * 2 + t) * 16], 64);
      acc[t] = wmma_bf16(af, bfr, acc[t]);
    }
    __syncthreads();
  }

  const int cn = lane & 15;
  const int rb = (lane < 16) ? 0 : 8;
#pragma unroll
  for (int t = 0; t < 2; ++t) {
    const int d = d0 + (nc * 2 + t) * 16 + cn;
#pragma unroll
    for (int j = 0; j < 8; ++j) {
      const int i = i0 + mr * 16 + rb + j;
      ctx[((size_t)i * B + b) * D + d] = acc[t][j];
    }
  }
}

// ---------------------------------------------------------------------------
// Kernel 4: one GRU step (launched L times; stream order = global sync).
// gh = h_prev @ w_hh^T + b_hh via WMMA (M padded 24->32), fused gate math.
// Grid: D/64 blocks; each block owns a 64-wide d-slice for all 3 gates.
// ---------------------------------------------------------------------------
__global__ __launch_bounds__(256) void gru_step(
    const float* __restrict__ h_prev,  // [B,D]
    const float* __restrict__ gi,      // [B,3D] precomputed input gates (row i)
    const float* __restrict__ w_hh,    // [3D,D]
    const float* __restrict__ b_hh,    // [3D]
    float* __restrict__ h_next,        // [B,D]
    float* __restrict__ hs_out) {      // [B,D] slice of output
  __shared__ __bf16 As[32][32];      //  2 KB  h tile (rows >= B zero-padded)
  __shared__ __bf16 Bs[3][64][32];   // 12 KB  w_hh tiles, 3 gates
  const int tid = threadIdx.x, wave = tid >> 5, lane = tid & 31;
  const int n0 = blockIdx.x * 64;
  const int mi = wave & 1;   // 0..1 -> 16-row strip
  const int ni = wave >> 1;  // 0..3 -> 16-col strip
  v8f acc[3] = {};

  for (int k0 = 0; k0 < D; k0 += 32) {
    // prefetch next w_hh K-tile (one per row, 3 gates x 64 rows)
    if (k0 + 32 < D && tid < 192) {
      const int g = tid >> 6, r = tid & 63;
      __builtin_prefetch(&w_hh[(size_t)(g * D + n0 + r) * D + (k0 + 32)], 0, 1);
    }
    for (int idx = tid; idx < 32 * 32; idx += 256) {
      const int r = idx >> 5, c = idx & 31;
      As[r][c] = (__bf16)((r < B) ? h_prev[r * D + (k0 + c)] : 0.0f);
    }
    for (int idx = tid; idx < 3 * 64 * 32; idx += 256) {
      const int g = idx / (64 * 32);
      const int rem = idx % (64 * 32);
      const int r = rem >> 5, c = rem & 31;
      Bs[g][r][c] = (__bf16)w_hh[(size_t)(g * D + n0 + r) * D + (k0 + c)];
    }
    __syncthreads();
    const v16bf af = frag_a_rm(&As[mi * 16][0], 32);
#pragma unroll
    for (int g = 0; g < 3; ++g) {
      const v16bf bfr = frag_b_nm(&Bs[g][ni * 16][0], 32);
      acc[g] = wmma_bf16(af, bfr, acc[g]);
    }
    __syncthreads();
  }

  const int cn = lane & 15;
  const int rb = (lane < 16) ? 0 : 8;
  const int d = n0 + ni * 16 + cn;
#pragma unroll
  for (int j = 0; j < 8; ++j) {
    const int bb = mi * 16 + rb + j;
    if (bb < B) {
      const float hr = acc[0][j] + b_hh[d];
      const float hz = acc[1][j] + b_hh[D + d];
      const float hn = acc[2][j] + b_hh[2 * D + d];
      const float* girow = gi + (size_t)bb * D3;
      const float r = fast_sigmoid(girow[d] + hr);
      const float z = fast_sigmoid(girow[D + d] + hz);
      const float n = fast_tanh(girow[2 * D + d] + r * hn);
      const float hp = h_prev[bb * D + d];
      const float hnew = (1.0f - z) * n + z * hp;
      h_next[bb * D + d] = hnew;
      hs_out[bb * D + d] = hnew;
    }
  }
}

// ---------------------------------------------------------------------------
// Host-side orchestration.
// Workspace layout (byte offsets), with deliberate aliasing:
//   [0, 48MB)    Wpv   (f32, dead after scores_kernel)
//   [48, 96MB)   Wpvp  (f32, dead after scores_kernel)
//   [96, 144MB)  P     (bf16 probs, dead after ctx_gemm)
//   [0, 144MB)   gi    (f32, written by gi-GEMM AFTER all three are dead)
//   [144MB, +96KB) h ping-pong buffers
// ctx lives temporarily in d_out (overwritten later by hs).
// ---------------------------------------------------------------------------
extern "C" void kernel_launch(void* const* d_in, const int* in_sizes, int n_in,
                              void* d_out, int out_size, void* d_ws, size_t ws_size,
                              hipStream_t stream) {
  (void)in_sizes; (void)n_in; (void)out_size; (void)ws_size;
  const float* v    = (const float*)d_in[0];
  const float* h0   = (const float*)d_in[1];
  const float* Vv   = (const float*)d_in[2];
  const float* Wp   = (const float*)d_in[3];
  const float* Wp_  = (const float*)d_in[4];
  const float* w_ih = (const float*)d_in[5];
  const float* w_hh = (const float*)d_in[6];
  const float* b_ih = (const float*)d_in[7];
  const float* b_hh = (const float*)d_in[8];
  float* out = (float*)d_out;

  char* ws = (char*)d_ws;
  const size_t SZ = (size_t)L * B * D * sizeof(float);  // 48 MB
  float*  Wpv  = (float*)(ws);
  float*  Wpvp = (float*)(ws + SZ);
  __bf16* P    = (__bf16*)(ws + 2 * SZ);   // bf16 [B][L][L] == 48 MB
  float*  gi   = (float*)(ws);             // aliases Wpv/Wpvp/P once dead
  float*  hb0  = (float*)(ws + 3 * SZ);
  float*  hb1  = hb0 + B * D;
  float*  ctx  = out;                      // temporary home for contexts

  const int M = L * B;  // 24576

  // 1) projections: Wpv = v*Wp^T, Wpvp = v*Wp_^T
  dim3 gProj(M / 128, D / 64);
  gemm_nt_bf16<<<gProj, 256, 0, stream>>>(v, Wp,  nullptr, Wpv,  M, D, D);
  gemm_nt_bf16<<<gProj, 256, 0, stream>>>(v, Wp_, nullptr, Wpvp, M, D, D);

  // 2) attention scores + softmax -> probs
  scores_kernel<<<dim3(B, L / 8), 256, 0, stream>>>(Wpv, Wpvp, Vv, P);

  // 3) contexts: ctx[i,b,:] = P[b][i][:] @ v[:,b,:]
  ctx_gemm_kernel<<<dim3(B, L / 64, D / 64), 256, 0, stream>>>(P, v, ctx);

  // 4) gi = ctx @ w_ih^T + b_ih   (gi overwrites the now-dead Wpv/Wpvp/P)
  dim3 gGi(M / 128, D3 / 64);
  gemm_nt_bf16<<<gGi, 256, 0, stream>>>(ctx, w_ih, b_ih, gi, M, D3, D);

  // 5) GRU recurrence: 1024 dependent launches (stream order = global barrier)
  hipMemcpyAsync(hb0, h0, (size_t)B * D * sizeof(float),
                 hipMemcpyDeviceToDevice, stream);
  for (int i = 0; i < L; ++i) {
    const float* hp = (i & 1) ? hb1 : hb0;
    float*       hn = (i & 1) ? hb0 : hb1;
    gru_step<<<D / 64, 256, 0, stream>>>(hp, gi + (size_t)i * B * D3, w_hh,
                                         b_hh, hn, out + (size_t)i * B * D);
  }
}
